// RepresentationFusion_41171556500075
// MI455X (gfx1250) — matrix operand order repaired and checked by reading, not compiled, not run on hardware
//
#include <hip/hip_runtime.h>
#include <hip/hip_bf16.h>

typedef __attribute__((ext_vector_type(16))) __bf16 v16bf;
typedef __attribute__((ext_vector_type(8)))  __bf16 v8bf;
typedef __attribute__((ext_vector_type(4)))  __bf16 v4bf;
typedef __attribute__((ext_vector_type(8)))  float  v8f;

constexpr int kS  = 512;
constexpr int kB  = 64;
constexpr int kT  = 768;
constexpr int kA  = 512;
constexpr int kH1 = 768;
constexpr int kH2 = 640;
constexpr int kN  = kH1 + kH2;     // 1408 concatenated output cols
constexpr int kSB = kS * kB;       // 32768 rows
constexpr int kKT = kT / 32;       // 24 K-tiles of 32
constexpr int kNT = kN / 16;       // 88 N-tiles of 16
constexpr int kPASS = kNT / 8;     // 11 N-tiles per wave

__device__ __forceinline__ void split_bf16(float x, __bf16& h, __bf16& l) {
    __bf16 hh = (__bf16)x;
    h = hh;
    l = (__bf16)(x - (float)hh);
}

// ---------------------------------------------------------------------------
// Pack W11[:T] || W12[:T] into WMMA-B per-lane layout, bf16 hi/lo pair.
// Linear index = ((kt*kNT + nt)*32 + lane)*16 + j
// lane: col n = nt*16 + (lane&15); K = kt*32 + (lane>>4)*16 + j
// ---------------------------------------------------------------------------
__global__ void pack_w_kernel(const float* __restrict__ W11,
                              const float* __restrict__ W12,
                              __bf16* __restrict__ Bhi, __bf16* __restrict__ Blo) {
    int idx = blockIdx.x * blockDim.x + threadIdx.x;
    if (idx >= kKT * kNT * 512) return;
    int j    = idx & 15;
    int lane = (idx >> 4) & 31;
    int knt  = idx >> 9;
    int nt   = knt % kNT;
    int kt   = knt / kNT;
    int k = kt * 32 + (lane >> 4) * 16 + j;
    int n = nt * 16 + (lane & 15);
    float x = (n < kH1) ? W11[(size_t)k * kH1 + n]
                        : W12[(size_t)k * kH2 + (n - kH1)];
    __bf16 h, l;
    split_bf16(x, h, l);
    Bhi[idx] = h;
    Blo[idx] = l;
}

__global__ void pack_v_kernel(const float* __restrict__ W21,
                              const float* __restrict__ W22,
                              float* __restrict__ vcat) {
    int n = blockIdx.x * blockDim.x + threadIdx.x;
    if (n >= kN) return;
    vcat[n] = (n < kH1) ? W21[n] : W22[n - kH1];
}

// biascat[b, n] = (text @ W11[T:])[b,n] + b11[n]  (n < H1)
//              = (anp  @ W12[T:])[b,n-H1] + b12   (n >= H1)
// b21/b22 dropped: constant along softmax axis.
__global__ void bias_kernel(const float* __restrict__ text,
                            const float* __restrict__ anp,
                            const float* __restrict__ W11,
                            const float* __restrict__ b11,
                            const float* __restrict__ W12,
                            const float* __restrict__ b12,
                            float* __restrict__ biascat) {
    int idx = blockIdx.x * blockDim.x + threadIdx.x;
    if (idx >= kB * kN) return;
    int n = idx % kN;
    int b = idx / kN;
    float acc;
    if (n < kH1) {
        acc = b11[n];
        const float* w = W11 + (size_t)kT * kH1 + n;   // rows T..2T-1
        const float* x = text + (size_t)b * kT;
        for (int t = 0; t < kT; ++t) acc += x[t] * w[(size_t)t * kH1];
    } else {
        int n2 = n - kH1;
        acc = b12[n2];
        const float* w = W12 + (size_t)kT * kH2 + n2;  // rows T..T+A-1
        const float* x = anp + (size_t)b * kA;
        for (int t = 0; t < kA; ++t) acc += x[t] * w[(size_t)t * kH2];
    }
    biascat[idx] = acc;
}

// ---------------------------------------------------------------------------
// Fused GEMM + tanh + score reduction.
// One workgroup (8 waves) owns a 16-row block of text_seq (staged once in LDS
// as bf16 hi/lo). Loop nest: outer K (24 steps), inner 11 N-tiles per wave
// with 11 independent f32 accumulators -> 33 WMMAs in 11 independent chains per
// K-step (keeps the XDL pipe full), A fragments loaded from LDS once per
// K-step, B tiles loaded with immediate offsets off a per-wave base pointer.
// bf16x3 compensation (Ahi*Bhi + Ahi*Blo + Alo*Bhi) recovers ~fp32 accuracy.
// ---------------------------------------------------------------------------
__global__ __launch_bounds__(256)
void fused_gemm_score_kernel(const float* __restrict__ tseq,
                             const __bf16* __restrict__ Bhi,
                             const __bf16* __restrict__ Blo,
                             const float* __restrict__ biascat,
                             const float* __restrict__ vcat,
                             float* __restrict__ s1,
                             float* __restrict__ s2) {
    __shared__ __align__(16) __bf16 Ahi[16][776];  // pad 768->776 (16B-aligned rows)
    __shared__ __align__(16) __bf16 Alo[16][776];
    __shared__ float Rs[2][8][16];                 // [target][wave][row]

    const int tid   = threadIdx.x;
    const int lane  = tid & 31;
    const int wave  = tid >> 5;
    const int lrow  = lane & 15;    // A row / C column within tile
    const int khalf = lane >> 4;    // K-half selector
    const int rowbase = blockIdx.x * 16;

    // Stage the 16x768 f32 A block once, converted to bf16 hi/lo.
    for (int e = tid; e < 16 * 192; e += 256) {
        int row = e / 192;
        int c4  = (e % 192) * 4;
        const float4 av = *(const float4*)(tseq + (size_t)(rowbase + row) * kT + c4);
        v4bf hv, lv;
        __bf16 h, l;
        split_bf16(av.x, h, l); hv[0] = h; lv[0] = l;
        split_bf16(av.y, h, l); hv[1] = h; lv[1] = l;
        split_bf16(av.z, h, l); hv[2] = h; lv[2] = l;
        split_bf16(av.w, h, l); hv[3] = h; lv[3] = l;
        *(v4bf*)&Ahi[row][c4] = hv;
        *(v4bf*)&Alo[row][c4] = lv;
    }
    __syncthreads();

    // 11 independent accumulators: tile nt = p*8 + wave.
    v8f acc[kPASS];
    v8f vzero = {};
#pragma unroll
    for (int p = 0; p < kPASS; ++p) acc[p] = vzero;

    // Per-wave B base: element offset nt*512 + lane*16 with nt = p*8+wave.
    // p stride = 4096 elements (8 KB), kt stride = 45056 elements (88 KB).
    const unsigned lanebase = (unsigned)wave * 512u + (unsigned)lane * 16u;

    for (int kt = 0; kt < kKT; ++kt) {
        const int kb = kt * 32 + khalf * 16;
        v8bf a0h = *(const v8bf*)&Ahi[lrow][kb];
        v8bf a1h = *(const v8bf*)&Ahi[lrow][kb + 8];
        v8bf a0l = *(const v8bf*)&Alo[lrow][kb];
        v8bf a1l = *(const v8bf*)&Alo[lrow][kb + 8];
        v16bf ah = __builtin_shufflevector(a0h, a1h,
            0, 1, 2, 3, 4, 5, 6, 7, 8, 9, 10, 11, 12, 13, 14, 15);
        v16bf al = __builtin_shufflevector(a0l, a1l,
            0, 1, 2, 3, 4, 5, 6, 7, 8, 9, 10, 11, 12, 13, 14, 15);

        const __bf16* bkh = Bhi + lanebase + (unsigned)kt * 45056u;
        const __bf16* bkl = Blo + lanebase + (unsigned)kt * 45056u;
#pragma unroll
        for (int p = 0; p < kPASS; ++p) {
            v16bf bh = *(const v16bf*)(bkh + p * 4096);  // imm-offset loads
            v16bf bl = *(const v16bf*)(bkl + p * 4096);
            acc[p] = __builtin_amdgcn_wmma_f32_16x16x32_bf16(
                false, ah, false, bh, (short)0, acc[p], false, false);
            acc[p] = __builtin_amdgcn_wmma_f32_16x16x32_bf16(
                false, ah, false, bl, (short)0, acc[p], false, false);
            acc[p] = __builtin_amdgcn_wmma_f32_16x16x32_bf16(
                false, al, false, bh, (short)0, acc[p], false, false);
        }
    }

    // Epilogue: tanh(z + bias) * vcat, reduce over all N into per-row sums.
    float raccA[8];   // rows -> s1 (N-tiles 0..47  <=> p < 6)
    float raccB[8];   // rows -> s2 (N-tiles 48..87 <=> p >= 6)
#pragma unroll
    for (int i = 0; i < 8; ++i) { raccA[i] = 0.f; raccB[i] = 0.f; }

#pragma unroll
    for (int p = 0; p < kPASS; ++p) {
        const int nt = p * 8 + wave;
        const int ncol = nt * 16 + lrow;
        const float v = vcat[ncol];
#pragma unroll
        for (int i = 0; i < 8; ++i) {
            const int row = khalf * 8 + i;                 // C layout row
            const int bidx = (rowbase + row) & (kB - 1);   // batch index
            float x = acc[p][i] + biascat[(size_t)bidx * kN + ncol];
            float contrib = tanhf(x) * v;
            contrib += __shfl_xor(contrib, 1, 32);
            contrib += __shfl_xor(contrib, 2, 32);
            contrib += __shfl_xor(contrib, 4, 32);
            contrib += __shfl_xor(contrib, 8, 32);         // sum over 16 cols
            if (p < 6) raccA[i] += contrib; else raccB[i] += contrib;
        }
    }

    // Deterministic cross-wave reduction via LDS (no atomics).
    if (lrow == 0) {
#pragma unroll
        for (int i = 0; i < 8; ++i) {
            Rs[0][wave][khalf * 8 + i] = raccA[i];
            Rs[1][wave][khalf * 8 + i] = raccB[i];
        }
    }
    __syncthreads();
    if (tid < 32) {
        const int tgt = tid >> 4;
        const int row = tid & 15;
        float s = 0.f;
#pragma unroll
        for (int w = 0; w < 8; ++w) s += Rs[tgt][w][row];
        (tgt ? s2 : s1)[rowbase + row] = s;
    }
}

// ---------------------------------------------------------------------------
// Dual softmax over S per batch column; emits combined coefficient
// coef[s,b] = 0.5/S * (softmax(s1)[s,b] + softmax(s2)[s,b])
// ---------------------------------------------------------------------------
__global__ __launch_bounds__(256)
void softmax_kernel(const float* __restrict__ s1, const float* __restrict__ s2,
                    float* __restrict__ coef) {
    __shared__ float red[256];
    const int b = blockIdx.x;
    const int tid = threadIdx.x;
    const float v1a = s1[(size_t)tid * kB + b];
    const float v1b = s1[(size_t)(tid + 256) * kB + b];
    const float v2a = s2[(size_t)tid * kB + b];
    const float v2b = s2[(size_t)(tid + 256) * kB + b];

    red[tid] = fmaxf(v1a, v1b); __syncthreads();
    for (int o = 128; o > 0; o >>= 1) {
        if (tid < o) red[tid] = fmaxf(red[tid], red[tid + o]);
        __syncthreads();
    }
    const float m1 = red[0]; __syncthreads();

    red[tid] = fmaxf(v2a, v2b); __syncthreads();
    for (int o = 128; o > 0; o >>= 1) {
        if (tid < o) red[tid] = fmaxf(red[tid], red[tid + o]);
        __syncthreads();
    }
    const float m2 = red[0]; __syncthreads();

    const float e1a = expf(v1a - m1), e1b = expf(v1b - m1);
    const float e2a = expf(v2a - m2), e2b = expf(v2b - m2);

    red[tid] = e1a + e1b; __syncthreads();
    for (int o = 128; o > 0; o >>= 1) {
        if (tid < o) red[tid] += red[tid + o];
        __syncthreads();
    }
    const float inv1 = 1.f / red[0]; __syncthreads();

    red[tid] = e2a + e2b; __syncthreads();
    for (int o = 128; o > 0; o >>= 1) {
        if (tid < o) red[tid] += red[tid + o];
        __syncthreads();
    }
    const float inv2 = 1.f / red[0];

    const float sc = 0.5f / (float)kS;
    coef[(size_t)tid * kB + b]         = (e1a * inv1 + e2a * inv2) * sc;
    coef[(size_t)(tid + 256) * kB + b] = (e1b * inv1 + e2b * inv2) * sc;
}

// out[b,t] = sum_s coef[s,b] * text_seq[s,b,t]
__global__ __launch_bounds__(256)
void out_kernel(const float* __restrict__ tseq, const float* __restrict__ coef,
                float* __restrict__ out) {
    __shared__ float cf[kS];
    const int b = blockIdx.x;
    const int tid = threadIdx.x;
    const int t = blockIdx.y * 256 + tid;
    for (int s = tid; s < kS; s += 256) cf[s] = coef[(size_t)s * kB + b];
    __syncthreads();
    float acc = 0.f;
    const float* base = tseq + (size_t)b * kT + t;
#pragma unroll 4
    for (int s = 0; s < kS; ++s) acc += cf[s] * base[(size_t)s * kB * kT];
    out[(size_t)b * kT + t] = acc;
}

extern "C" void kernel_launch(void* const* d_in, const int* in_sizes, int n_in,
                              void* d_out, int out_size, void* d_ws, size_t ws_size,
                              hipStream_t stream) {
    (void)in_sizes; (void)n_in; (void)out_size; (void)ws_size;
    const float* text = (const float*)d_in[0];
    const float* anp  = (const float*)d_in[1];
    const float* tseq = (const float*)d_in[2];
    const float* W11  = (const float*)d_in[3];
    const float* b11  = (const float*)d_in[4];
    const float* W21  = (const float*)d_in[5];
    // d_in[6] = b21: constant along softmax axis -> cancels
    const float* W12  = (const float*)d_in[7];
    const float* b12  = (const float*)d_in[8];
    const float* W22  = (const float*)d_in[9];
    // d_in[10] = b22: cancels
    float* out = (float*)d_out;

    char* p = (char*)d_ws;
    auto take = [&p](size_t bytes) -> char* {
        char* q = p;
        p += (bytes + 255) & ~(size_t)255;
        return q;
    };
    float*  biascat = (float*) take((size_t)kB * kN * sizeof(float));
    float*  vcat    = (float*) take((size_t)kN * sizeof(float));
    float*  s1      = (float*) take((size_t)kSB * sizeof(float));
    float*  s2      = (float*) take((size_t)kSB * sizeof(float));
    float*  coef    = (float*) take((size_t)kSB * sizeof(float));
    __bf16* Bhi     = (__bf16*)take((size_t)kKT * kNT * 512 * sizeof(__bf16));
    __bf16* Blo     = (__bf16*)take((size_t)kKT * kNT * 512 * sizeof(__bf16));

    pack_w_kernel<<<(kKT * kNT * 512 + 255) / 256, 256, 0, stream>>>(W11, W12, Bhi, Blo);
    pack_v_kernel<<<(kN + 255) / 256, 256, 0, stream>>>(W21, W22, vcat);
    bias_kernel<<<(kB * kN + 255) / 256, 256, 0, stream>>>(text, anp, W11, b11, W12, b12, biascat);
    fused_gemm_score_kernel<<<dim3(kSB / 16, 1, 1), 256, 0, stream>>>(
        tseq, Bhi, Blo, biascat, vcat, s1, s2);
    softmax_kernel<<<kB, 256, 0, stream>>>(s1, s2, coef);
    out_kernel<<<dim3(kB, kT / 256, 1), 256, 0, stream>>>(tseq, coef, out);
}
